// FPModule_29729763623229
// MI455X (gfx1250) — matrix operand order repaired and checked by reading, not compiled, not run on hardware
//
#include <hip/hip_runtime.h>
#include <hip/hip_bf16.h>
#include <stdint.h>

#define B_      16
#define NC_     1024
#define NF_     4096
#define C_IN_   256
#define C_SKIP_ 128
#define C_HID_  256
#define D_IN_   384           // C_IN + C_SKIP
#define M_TOT   (B_ * NF_)    // 65536 rows
#define BN_EPS_ 1e-5f

typedef __attribute__((ext_vector_type(16))) __bf16 v16bf;
typedef __attribute__((ext_vector_type(8)))  float  v8f;

__device__ __forceinline__ unsigned short f2bf(float f) {
  uint32_t u = __float_as_uint(f);
  u += 0x7FFFu + ((u >> 16) & 1u);         // round-to-nearest-even
  return (unsigned short)(u >> 16);
}
__device__ __forceinline__ float bf2f(unsigned short h) {
  return __uint_as_float(((uint32_t)h) << 16);
}

// ---------------------------------------------------------------- KNN top-3
__global__ __launch_bounds__(256)
void knn_kernel(const float* __restrict__ pos, const float* __restrict__ pos_skip,
                int* __restrict__ idx_out, float* __restrict__ w_out) {
  __shared__ float spc[NC_ * 3];                       // 12 KB coarse positions
  const int chunks = NF_ / 256;
  const int b = blockIdx.x / chunks;
  const int chunk = blockIdx.x % chunks;
  const float* pc = pos + (size_t)b * NC_ * 3;
  for (int i = threadIdx.x; i < NC_ * 3; i += 256) spc[i] = pc[i];
  __syncthreads();

  const size_t gf = (size_t)b * NF_ + chunk * 256 + threadIdx.x;
  const float px = pos_skip[gf * 3 + 0];
  const float py = pos_skip[gf * 3 + 1];
  const float pz = pos_skip[gf * 3 + 2];

  float d0 = 3.4e38f, d1 = 3.4e38f, d2 = 3.4e38f;
  int   i0 = 0, i1 = 0, i2 = 0;
  for (int j = 0; j < NC_; ++j) {
    float dx = spc[3 * j + 0] - px;
    float dy = spc[3 * j + 1] - py;
    float dz = spc[3 * j + 2] - pz;
    float d = dx * dx + dy * dy + dz * dz;
    if (d < d0)      { d2 = d1; i2 = i1; d1 = d0; i1 = i0; d0 = d; i0 = j; }
    else if (d < d1) { d2 = d1; i2 = i1; d1 = d;  i1 = j; }
    else if (d < d2) { d2 = d;  i2 = j; }
  }
  idx_out[gf * 3 + 0] = b * NC_ + i0;
  idx_out[gf * 3 + 1] = b * NC_ + i1;
  idx_out[gf * 3 + 2] = b * NC_ + i2;
  w_out[gf * 3 + 0] = 1.0f / fmaxf(d0, 1e-16f);
  w_out[gf * 3 + 1] = 1.0f / fmaxf(d1, 1e-16f);
  w_out[gf * 3 + 2] = 1.0f / fmaxf(d2, 1e-16f);
}

// --------------------------------------------- gather + concat -> bf16 h
__global__ __launch_bounds__(256)
void interp_kernel(const float* __restrict__ x, const float* __restrict__ x_skip,
                   const int* __restrict__ idx, const float* __restrict__ wts,
                   unsigned short* __restrict__ h) {
  const size_t f = blockIdx.x;
  const int c = threadIdx.x;
  const int   i0 = idx[f * 3 + 0], i1 = idx[f * 3 + 1], i2 = idx[f * 3 + 2];
  const float w0 = wts[f * 3 + 0], w1 = wts[f * 3 + 1], w2 = wts[f * 3 + 2];
  const float inv = 1.0f / (w0 + w1 + w2);
  float v = (w0 * x[(size_t)i0 * C_IN_ + c] +
             w1 * x[(size_t)i1 * C_IN_ + c] +
             w2 * x[(size_t)i2 * C_IN_ + c]) * inv;
  h[f * D_IN_ + c] = f2bf(v);
  if (c < C_SKIP_) h[f * D_IN_ + C_IN_ + c] = f2bf(x_skip[f * C_SKIP_ + c]);
}

// ---------------------------------------- W1 (K x N) -> Wt1 (N x K) bf16
__global__ __launch_bounds__(256)
void wt1_kernel(const float* __restrict__ W1, unsigned short* __restrict__ Wt1) {
  const int i = blockIdx.x * 256 + threadIdx.x;      // over 384*256
  const int k = i / C_HID_, n = i % C_HID_;
  Wt1[(size_t)n * D_IN_ + k] = f2bf(W1[i]);
}

// ------------------------------------------------- WMMA GEMM + bias + ReLU
// A: M x K bf16 row-major.  Wt: N x K bf16 (pre-transposed).  N fixed = 256.
// Wave computes a 16x64 tile: one A fragment reused across 4 B tiles.
template <int K>
__global__ __launch_bounds__(256)
void gemm_relu_kernel(const unsigned short* __restrict__ A,
                      const unsigned short* __restrict__ Wt,
                      const float* __restrict__ bias,
                      unsigned short* __restrict__ outB,
                      float* __restrict__ outF,
                      int out_f32) {
  const int tid  = threadIdx.x;
  const int wave = tid >> 5;
  const int lane = tid & 31;
  const int hh   = lane >> 4;          // half-wave id
  const int nl   = lane & 15;
  const int mbase = (blockIdx.x * 2 + (wave >> 2)) * 16;
  const int nbase = (wave & 3) * 64;

  const unsigned short* arow = A  + (size_t)(mbase + nl) * K;
  const unsigned short* br0  = Wt + (size_t)(nbase +  0 + nl) * K;
  const unsigned short* br1  = Wt + (size_t)(nbase + 16 + nl) * K;
  const unsigned short* br2  = Wt + (size_t)(nbase + 32 + nl) * K;
  const unsigned short* br3  = Wt + (size_t)(nbase + 48 + nl) * K;

  v8f acc0 = {}, acc1 = {}, acc2 = {}, acc3 = {};

#pragma unroll
  for (int k = 0; k < K; k += 32) {
    union { uint4 q[2]; v16bf v; } a, b0, b1, b2, b3;
    // A frag (16-bit 16x32): lane holds contiguous K run [k+16h..k+16h+15]
    a.q[0] = *(const uint4*)(arow + k + 16 * hh);
    a.q[1] = *(const uint4*)(arow + k + 16 * hh + 8);
    // B frag (32x16): lane holds contiguous K run [k+16h..k+16h+15]
    b0.q[0] = *(const uint4*)(br0 + k + 16 * hh);
    b0.q[1] = *(const uint4*)(br0 + k + 16 * hh + 8);
    b1.q[0] = *(const uint4*)(br1 + k + 16 * hh);
    b1.q[1] = *(const uint4*)(br1 + k + 16 * hh + 8);
    b2.q[0] = *(const uint4*)(br2 + k + 16 * hh);
    b2.q[1] = *(const uint4*)(br2 + k + 16 * hh + 8);
    b3.q[0] = *(const uint4*)(br3 + k + 16 * hh);
    b3.q[1] = *(const uint4*)(br3 + k + 16 * hh + 8);

    acc0 = __builtin_amdgcn_wmma_f32_16x16x32_bf16(false, a.v, false, b0.v,
                                                   (short)0, acc0, false, false);
    acc1 = __builtin_amdgcn_wmma_f32_16x16x32_bf16(false, a.v, false, b1.v,
                                                   (short)0, acc1, false, false);
    acc2 = __builtin_amdgcn_wmma_f32_16x16x32_bf16(false, a.v, false, b2.v,
                                                   (short)0, acc2, false, false);
    acc3 = __builtin_amdgcn_wmma_f32_16x16x32_bf16(false, a.v, false, b3.v,
                                                   (short)0, acc3, false, false);
  }

  const float bv0 = bias[nbase +  0 + nl];
  const float bv1 = bias[nbase + 16 + nl];
  const float bv2 = bias[nbase + 32 + nl];
  const float bv3 = bias[nbase + 48 + nl];

#pragma unroll
  for (int r = 0; r < 8; ++r) {
    const int m = mbase + r + 8 * hh;            // C/D layout: VGPR r -> M=r+8h
    const size_t ro = (size_t)m * C_HID_;
    float v0 = fmaxf(acc0[r] + bv0, 0.0f);
    float v1 = fmaxf(acc1[r] + bv1, 0.0f);
    float v2 = fmaxf(acc2[r] + bv2, 0.0f);
    float v3 = fmaxf(acc3[r] + bv3, 0.0f);
    if (out_f32) {
      outF[ro + nbase +  0 + nl] = v0;
      outF[ro + nbase + 16 + nl] = v1;
      outF[ro + nbase + 32 + nl] = v2;
      outF[ro + nbase + 48 + nl] = v3;
    } else {
      outB[ro + nbase +  0 + nl] = f2bf(v0);
      outB[ro + nbase + 16 + nl] = f2bf(v1);
      outB[ro + nbase + 32 + nl] = f2bf(v2);
      outB[ro + nbase + 48 + nl] = f2bf(v3);
    }
  }
}

// -------------------------------- deterministic per-channel sum / sumsq
__global__ __launch_bounds__(256)
void colsum_bf16_kernel(const unsigned short* __restrict__ z,
                        float* __restrict__ sum, float* __restrict__ sq) {
  const int n = blockIdx.x, t = threadIdx.x;
  float s = 0.f, s2 = 0.f;
  for (int r = t; r < M_TOT; r += 256) {
    float v = bf2f(z[(size_t)r * C_HID_ + n]);
    s += v; s2 += v * v;
  }
  __shared__ float ss[256], sq2[256];
  ss[t] = s; sq2[t] = s2; __syncthreads();
  for (int o = 128; o > 0; o >>= 1) {
    if (t < o) { ss[t] += ss[t + o]; sq2[t] += sq2[t + o]; }
    __syncthreads();
  }
  if (t == 0) { sum[n] = ss[0]; sq[n] = sq2[0]; }
}

__global__ __launch_bounds__(256)
void colsum_f32_kernel(const float* __restrict__ z,
                       float* __restrict__ sum, float* __restrict__ sq) {
  const int n = blockIdx.x, t = threadIdx.x;
  float s = 0.f, s2 = 0.f;
  for (int r = t; r < M_TOT; r += 256) {
    float v = z[(size_t)r * C_HID_ + n];
    s += v; s2 += v * v;
  }
  __shared__ float ss[256], sq2[256];
  ss[t] = s; sq2[t] = s2; __syncthreads();
  for (int o = 128; o > 0; o >>= 1) {
    if (t < o) { ss[t] += ss[t + o]; sq2[t] += sq2[t + o]; }
    __syncthreads();
  }
  if (t == 0) { sum[n] = ss[0]; sq[n] = sq2[0]; }
}

// ---- fold BN1 affine into W2 / b2:  a1 = z1*s + t ;  a1@W2 = z1@(sW2)+t@W2
__global__ __launch_bounds__(256)
void fold_kernel(const float* __restrict__ W2, const float* __restrict__ b2,
                 const float* __restrict__ g1, const float* __restrict__ be1,
                 const float* __restrict__ sum1, const float* __restrict__ sq1,
                 unsigned short* __restrict__ Wt2, float* __restrict__ b2f) {
  const int n = blockIdx.x, k = threadIdx.x;
  const float inv_m = 1.0f / (float)M_TOT;
  const float mean = sum1[k] * inv_m;
  const float var  = sq1[k] * inv_m - mean * mean;
  const float s    = g1[k] * rsqrtf(var + BN_EPS_);
  const float t    = be1[k] - mean * s;
  const float w    = W2[(size_t)k * C_HID_ + n];
  Wt2[(size_t)n * C_HID_ + k] = f2bf(s * w);      // transposed + scaled
  __shared__ float red[256];
  red[k] = t * w; __syncthreads();
  for (int o = 128; o > 0; o >>= 1) {
    if (k < o) red[k] += red[k + o];
    __syncthreads();
  }
  if (k == 0) b2f[n] = b2[n] + red[0];
}

// ------------------------------- final BN2 affine, in place on d_out
__global__ __launch_bounds__(256)
void affine_kernel(float* __restrict__ out, const float* __restrict__ g2,
                   const float* __restrict__ be2, const float* __restrict__ sum2,
                   const float* __restrict__ sq2) {
  const size_t i = (size_t)blockIdx.x * 256 + threadIdx.x;
  const int n = (int)(i & (C_HID_ - 1));
  const float inv_m = 1.0f / (float)M_TOT;
  const float mean = sum2[n] * inv_m;
  const float var  = sq2[n] * inv_m - mean * mean;
  const float s    = g2[n] * rsqrtf(var + BN_EPS_);
  const float t    = be2[n] - mean * s;
  out[i] = out[i] * s + t;
}

extern "C" void kernel_launch(void* const* d_in, const int* in_sizes, int n_in,
                              void* d_out, int out_size, void* d_ws, size_t ws_size,
                              hipStream_t stream) {
  (void)in_sizes; (void)n_in; (void)out_size; (void)ws_size;
  const float* x        = (const float*)d_in[0];
  const float* pos      = (const float*)d_in[1];
  /* batch (int64)      d_in[2] -- derived analytically */
  const float* x_skip   = (const float*)d_in[3];
  const float* pos_skip = (const float*)d_in[4];
  /* batch_skip         d_in[5] -- derived analytically */
  const float* W1  = (const float*)d_in[6];
  const float* b1  = (const float*)d_in[7];
  const float* g1  = (const float*)d_in[8];
  const float* be1 = (const float*)d_in[9];
  const float* W2  = (const float*)d_in[10];
  const float* b2  = (const float*)d_in[11];
  const float* g2  = (const float*)d_in[12];
  const float* be2 = (const float*)d_in[13];
  float* out = (float*)d_out;

  char* ws = (char*)d_ws;
  size_t o = 0;
  unsigned short* h   = (unsigned short*)(ws + o); o += (size_t)M_TOT * D_IN_ * 2;   // 50.3 MB
  unsigned short* z1  = (unsigned short*)(ws + o); o += (size_t)M_TOT * C_HID_ * 2;  // 33.6 MB
  unsigned short* wt1 = (unsigned short*)(ws + o); o += (size_t)C_HID_ * D_IN_ * 2;
  unsigned short* wt2 = (unsigned short*)(ws + o); o += (size_t)C_HID_ * C_HID_ * 2;
  int*   knn_idx = (int*)(ws + o);   o += (size_t)M_TOT * 3 * 4;
  float* knn_w   = (float*)(ws + o); o += (size_t)M_TOT * 3 * 4;
  float* s1  = (float*)(ws + o); o += 1024;
  float* q1  = (float*)(ws + o); o += 1024;
  float* s2  = (float*)(ws + o); o += 1024;
  float* q2  = (float*)(ws + o); o += 1024;
  float* b2f = (float*)(ws + o); o += 1024;

  // 1) KNN (LDS-cached coarse positions), 2) gather+concat -> bf16
  knn_kernel<<<B_ * (NF_ / 256), 256, 0, stream>>>(pos, pos_skip, knn_idx, knn_w);
  interp_kernel<<<M_TOT, 256, 0, stream>>>(x, x_skip, knn_idx, knn_w, h);
  // 3) transpose-cast W1
  wt1_kernel<<<(D_IN_ * C_HID_) / 256, 256, 0, stream>>>(W1, wt1);
  // 4) layer-1 WMMA GEMM (M=65536, K=384, N=256), ReLU fused, z1 in bf16
  gemm_relu_kernel<D_IN_><<<M_TOT / 32, 256, 0, stream>>>(h, wt1, b1, z1, out, 0);
  // 5) BN1 stats, 6) fold BN1 into W2/b2
  colsum_bf16_kernel<<<C_HID_, 256, 0, stream>>>(z1, s1, q1);
  fold_kernel<<<C_HID_, 256, 0, stream>>>(W2, b2, g1, be1, s1, q1, wt2, b2f);
  // 7) layer-2 WMMA GEMM (K=256) -> f32 straight into d_out
  gemm_relu_kernel<C_HID_><<<M_TOT / 32, 256, 0, stream>>>(z1, wt2, b2f, z1, out, 1);
  // 8) BN2 stats, 9) in-place affine
  colsum_f32_kernel<<<C_HID_, 256, 0, stream>>>(out, s2, q2);
  affine_kernel<<<M_TOT, 256, 0, stream>>>(out, g2, be2, s2, q2);
}
